// VariationalMonteCarlo_69303592288695
// MI455X (gfx1250) — matrix operand order repaired and checked by reading, not compiled, not run on hardware
//
#include <hip/hip_runtime.h>
#include <hip/hip_bf16.h>

typedef __attribute__((ext_vector_type(16))) __bf16 v16bf;
typedef __attribute__((ext_vector_type(8)))  float  v8f;

#define NSITES 512   // N = 2*16*16
#define NHID   128   // NH
#define NSAMP  4096  // NS
#define NG     384   // 3*NH

__device__ __forceinline__ float fast_sigmoid(float x) {
    return __builtin_amdgcn_rcpf(1.0f + __expf(-x));     // v_rcp_f32, no IEEE divide
}
__device__ __forceinline__ float fast_tanh(float x) {
    return 1.0f - 2.0f * __builtin_amdgcn_rcpf(__expf(2.0f * x) + 1.0f);
}

// One workgroup = 16 samples, 8 wave32s. Wave w owns hidden columns [16w,16w+16).
// Recurrent weights live in VGPRs as WMMA B-fragments for all steps.
// Dense head computed by wave 0 as a 4-WMMA chain on the shared A-fragments.
__launch_bounds__(256, 1)
__global__ void gru_sample_kernel(
    const float* __restrict__ gk,    // gru_kernel  [2][384]
    const float* __restrict__ grk,   // gru_rkernel [128][384]
    const float* __restrict__ gb,    // gru_bias    [2][384]
    const float* __restrict__ dw,    // dense_w     [128][2]
    const float* __restrict__ db,    // dense_b     [2]
    const float* __restrict__ u,     // u           [4096][512]
    int*   __restrict__ out_s,       // samples     [4096][512] int32
    float* __restrict__ out_lp)      // log_prob    [4096]
{
    __shared__ __bf16 hbuf[2][16 * NHID];  // ping-pong h, row-major [m][k], bf16
    __shared__ float  redd[2][16];         // dense logits a0/a1 per sample row
    __shared__ int    s_word;              // 16 sample bits of previous site

    const int tid  = threadIdx.x;
    const int wave = tid >> 5;
    const int lane = tid & 31;
    const int n    = lane & 15;          // tile column (B/C/D) or tile row (A)
    const int hi   = lane >> 4;          // half-wave select
    const int koff = hi * 8;
    const int jcol = wave * 16 + n;      // this lane's hidden column (0..127)

    // ---- Recurrent weight B-fragments (resident in VGPRs) ----
    // B (32x16 bf16): lane n holds column n; lanes 0-15 K 0-7 & 16-23, lanes 16-31 K 8-15 & 24-31.
    v16bf Bf[3][4];
#pragma unroll
    for (int g = 0; g < 3; ++g) {
        const int col = g * NHID + jcol;
#pragma unroll
        for (int c = 0; c < 4; ++c) {
            v16bf f;
#pragma unroll
            for (int i = 0; i < 16; ++i) {
                const int k = c * 32 + koff + (i < 8 ? i : 8 + i);
                f[i] = (__bf16)grk[k * NG + col];
            }
            Bf[g][c] = f;
        }
    }
    // ---- Dense-head B-fragment: dense_w in columns 0/1, zero elsewhere ----
    v16bf Wf[4];
#pragma unroll
    for (int c = 0; c < 4; ++c) {
        v16bf f;
#pragma unroll
        for (int i = 0; i < 16; ++i) {
            const int k = c * 32 + koff + (i < 8 ? i : 8 + i);
            f[i] = (n < 2) ? (__bf16)dw[k * 2 + n] : (__bf16)0.0f;
        }
        Wf[c] = f;
    }

    // ---- Per-lane gate constants with b_in + b_rec folded in ----
    float cF[3], c0[3], c1[3];
#pragma unroll
    for (int g = 0; g < 3; ++g) {
        const int col = g * NHID + jcol;
        const float b = gb[col] + gb[NG + col];
        cF[g] = b;                      // x == 0 (t==0 only)
        c0[g] = gk[col]      + b;       // s_prev == 0
        c1[g] = gk[NG + col] + b;       // s_prev == 1
    }
    const float db0 = db[0];
    const float db1 = db[1];
    const v8f vzero = {0.f, 0.f, 0.f, 0.f, 0.f, 0.f, 0.f, 0.f};
    const int mg = blockIdx.x * 16 + lane;   // sample row (wave-0 sampling path)

    // ---- Peeled t = 0: h=0, x=0  =>  h_0 is a per-column constant ----
    float hprev[8];
    {
        const float z0  = fast_sigmoid(cF[0]);
        const float hc0 = fast_tanh(cF[2]);
        const float h0  = (1.0f - z0) * hc0;
#pragma unroll
        for (int r = 0; r < 8; ++r) hprev[r] = h0;
        const __bf16 h0b = (__bf16)h0;
#pragma unroll
        for (int r = 0; r < 8; ++r) hbuf[0][(r + koff) * NHID + jcol] = h0b;
    }
    if (tid == 0) s_word = 0;
    float logp = 0.0f;                 // wave 0, lanes 0..15
    int rd = 0;
    __syncthreads();

    // ---- Steady-state loop: iteration it consumes h_{it-1}, samples s_{it-1}, makes h_it ----
    for (int it = 1; it < NSITES; ++it) {
        // A-fragments of h_{it-1}.
        // A (16x32 bf16): lane row m = n; lanes 0-15 K 0-7 & 16-23, lanes 16-31 K 8-15 & 24-31.
        const __bf16* __restrict__ hb = &hbuf[rd][0];
        v16bf Af[4];
#pragma unroll
        for (int c = 0; c < 4; ++c) {
            v16bf f;
#pragma unroll
            for (int i = 0; i < 8; ++i) {
                f[i]     = hb[n * NHID + c * 32 + koff + i];
                f[i + 8] = hb[n * NHID + c * 32 + 16 + koff + i];
            }
            Af[c] = f;
        }

        // Gate GEMM (all waves): 3 gates x 4 K-chunks, C starts at inline 0
        v8f accz = __builtin_amdgcn_wmma_f32_16x16x32_bf16(false, Af[0], false, Bf[0][0],
                                                           (short)0, vzero, false, false);
        v8f accr = __builtin_amdgcn_wmma_f32_16x16x32_bf16(false, Af[0], false, Bf[1][0],
                                                           (short)0, vzero, false, false);
        v8f acch = __builtin_amdgcn_wmma_f32_16x16x32_bf16(false, Af[0], false, Bf[2][0],
                                                           (short)0, vzero, false, false);
#pragma unroll
        for (int c = 1; c < 4; ++c) {
            accz = __builtin_amdgcn_wmma_f32_16x16x32_bf16(false, Af[c], false, Bf[0][c],
                                                           (short)0, accz, false, false);
            accr = __builtin_amdgcn_wmma_f32_16x16x32_bf16(false, Af[c], false, Bf[1][c],
                                                           (short)0, accr, false, false);
            acch = __builtin_amdgcn_wmma_f32_16x16x32_bf16(false, Af[c], false, Bf[2][c],
                                                           (short)0, acch, false, false);
        }

        // Dense head + sampling of s_{it-1} (wave 0 only; EXEC all-ones for WMMA)
        if (wave == 0) {
            v8f accd = __builtin_amdgcn_wmma_f32_16x16x32_bf16(false, Af[0], false, Wf[0],
                                                               (short)0, vzero, false, false);
#pragma unroll
            for (int c = 1; c < 4; ++c)
                accd = __builtin_amdgcn_wmma_f32_16x16x32_bf16(false, Af[c], false, Wf[c],
                                                               (short)0, accd, false, false);
            // lanes n==0/1 hold logit columns 0/1 for rows m = koff..koff+7
            if (n < 2) {
#pragma unroll
                for (int r = 0; r < 8; ++r) redd[n][koff + r] = accd[r];
            }
            // same-wave LDS RAW: ordered by DScnt, no workgroup barrier needed
            if (lane < 16) {
                const int   ts = it - 1;
                const float a0 = redd[0][lane] + db0;
                const float a1 = redd[1][lane] + db1;
                const float p1 = __builtin_amdgcn_rcpf(1.0f + __expf(a0 - a1));
                const float um = u[mg * NSITES + ts];
                const int   s  = (um < p1) ? 1 : 0;
                const unsigned mask = (unsigned)__ballot(s != 0);
                if (lane == 0) s_word = (int)(mask & 0xFFFFu);
                out_s[mg * NSITES + ts] = s;
                logp += __logf((s ? p1 : (1.0f - p1)) + 1e-10f);
                if ((ts & 15) == 0)
                    __builtin_prefetch(u + mg * NSITES + ts + 16, 0, 0);
            }
        }
        __syncthreads();   // s_word published; everyone consumed hbuf[rd]

        // Gate nonlinearity + state update per C/D element (m = r + koff, col = jcol)
        const int sw = s_word;
        __bf16* __restrict__ hw = &hbuf[rd ^ 1][0];
#pragma unroll
        for (int r = 0; r < 8; ++r) {
            const int m   = r + koff;
            const int bit = (sw >> m) & 1;
            const float xz = bit ? c1[0] : c0[0];
            const float xr = bit ? c1[1] : c0[1];
            const float xh = bit ? c1[2] : c0[2];
            const float z  = fast_sigmoid(xz + accz[r]);
            const float rr = fast_sigmoid(xr + accr[r]);
            const float hc = fast_tanh(xh + rr * acch[r]);
            const float hn = hc + z * (hprev[r] - hc);
            hprev[r] = hn;
            hw[m * NHID + jcol] = (__bf16)hn;   // unique (m,jcol) per lane
        }
        __syncthreads();   // h_it complete before next iteration reads it
        rd ^= 1;
    }

    // ---- Epilogue: sample s_{511} from h_{511} ----
    if (wave == 0) {
        const __bf16* __restrict__ hb = &hbuf[rd][0];
        v16bf Af[4];
#pragma unroll
        for (int c = 0; c < 4; ++c) {
            v16bf f;
#pragma unroll
            for (int i = 0; i < 8; ++i) {
                f[i]     = hb[n * NHID + c * 32 + koff + i];
                f[i + 8] = hb[n * NHID + c * 32 + 16 + koff + i];
            }
            Af[c] = f;
        }
        v8f accd = __builtin_amdgcn_wmma_f32_16x16x32_bf16(false, Af[0], false, Wf[0],
                                                           (short)0, vzero, false, false);
#pragma unroll
        for (int c = 1; c < 4; ++c)
            accd = __builtin_amdgcn_wmma_f32_16x16x32_bf16(false, Af[c], false, Wf[c],
                                                           (short)0, accd, false, false);
        if (n < 2) {
#pragma unroll
            for (int r = 0; r < 8; ++r) redd[n][koff + r] = accd[r];
        }
        if (lane < 16) {
            const int   ts = NSITES - 1;
            const float a0 = redd[0][lane] + db0;
            const float a1 = redd[1][lane] + db1;
            const float p1 = __builtin_amdgcn_rcpf(1.0f + __expf(a0 - a1));
            const float um = u[mg * NSITES + ts];
            const int   s  = (um < p1) ? 1 : 0;
            out_s[mg * NSITES + ts] = s;
            logp += __logf((s ? p1 : (1.0f - p1)) + 1e-10f);
            out_lp[mg] = logp;
        }
    }
}

extern "C" void kernel_launch(void* const* d_in, const int* in_sizes, int n_in,
                              void* d_out, int out_size, void* d_ws, size_t ws_size,
                              hipStream_t stream) {
    (void)in_sizes; (void)n_in; (void)d_ws; (void)ws_size; (void)out_size;
    const float* gk  = (const float*)d_in[0];   // gru_kernel  [2][384]
    const float* grk = (const float*)d_in[1];   // gru_rkernel [128][384]
    const float* gb  = (const float*)d_in[2];   // gru_bias    [2][384]
    const float* dw  = (const float*)d_in[3];   // dense_w     [128][2]
    const float* db  = (const float*)d_in[4];   // dense_b     [2]
    const float* u   = (const float*)d_in[5];   // u           [4096][512]

    int*   out_s  = (int*)d_out;                       // samples, int32
    float* out_lp = (float*)d_out + NSAMP * NSITES;    // log_prob, float32

    dim3 grid(NSAMP / 16);   // 256 workgroups
    dim3 block(256);         // 8 wave32s
    gru_sample_kernel<<<grid, block, 0, stream>>>(gk, grk, gb, dw, db, u, out_s, out_lp);
}
